// ClusterNet_81192061763790
// MI455X (gfx1250) — matrix operand order repaired and checked by reading, not compiled, not run on hardware
//
#include <hip/hip_runtime.h>
#include <hip/hip_bf16.h>

// ClusterNet vote kernel for MI455X (gfx1250), wave32.
//
// out[l,p] = sum_c oc_c[p] * box51(oc_c*nn_l)[p] / box51(oc_c)[p]
// (the avg-pool "cnt" normalization cancels between numerator and denominator)
//
// Strategy: per-WG column-strip sweep. 72 vertical running sums (64 joint +
// 8 S planes, f32) live in LDS and are updated with one add/sub row per step.
// The 51-wide horizontal window is a banded 0/1 matrix multiply, done with
// 3x v_wmma_f32_16x16x32_f16 per (16-plane x 16-col) tile; 10 wave32 waves
// cover the 5x2 tile grid so EXEC is all ones at each WMMA.
// Combine stage factors w_c = oc_c/S_c once (256 divides -> 1 per thread),
// so the per-label reduction is pure FMA.

namespace {
constexpr int HH   = 1024;
constexpr int WW   = 1024;
constexpr int CC   = 8;
constexpr int LL   = 8;
constexpr int RAD  = 25;
constexpr int SW   = 32;              // output columns per strip
constexpr int XN   = SW + 2 * RAD;    // 82 valid x' entries per row
constexpr int XSP  = 96;              // staging row pitch
constexpr int XSTR = 132;             // VJS row pitch (bank-rotated, 16B aligned)
constexpr int NPL  = 80;              // 64 joint + 8 S + 8 zero-pad plane rows
constexpr int YB   = 128;             // output rows per workgroup
constexpr int TPB  = 320;             // 10 wave32 = 5 plane-tiles x 2 x-tiles
}

typedef __attribute__((ext_vector_type(16))) _Float16 v16h;
typedef __attribute__((ext_vector_type(8)))  float    v8f;

// Stage one (possibly OOB -> zero) row of all 8 oc planes (+EPS) and all 8 nn
// planes into LDS, columns x0-25 .. x0+56.
__device__ __forceinline__ void load_rows(const float* __restrict__ ca,
                                          const float* __restrict__ nnp,
                                          int yy, int x0,
                                          float (*oc)[XSP], float (*nl)[XSP],
                                          int tid) {
  const bool yok = (yy >= 0) && (yy < HH);
  for (int i = tid; i < CC * XN; i += TPB) {
    int p = i / XN;
    int s = i - p * XN;
    int g = x0 - RAD + s;
    bool ok = yok && ((unsigned)g < (unsigned)WW);
    float o = 0.0f, n = 0.0f;
    if (ok) {
      size_t off = (size_t)p * HH * WW + (size_t)yy * WW + (size_t)g;
      o = ca[off] + 1e-6f;   // EPS only on real pixels (zero-padded box sum)
      n = nnp[off];
    }
    oc[p][s] = o;
    nl[p][s] = n;
  }
}

// Vertical running-sum update: add incoming row products, optionally subtract
// outgoing row products. Planes 0..63: joint (p = c*8 + l); planes 64..71: S_c.
template <bool SUB>
__device__ __forceinline__ void vupdate(float* __restrict__ sVJS,
                                        const float (*ocN)[XSP], const float (*nnN)[XSP],
                                        const float (*ocO)[XSP], const float (*nnO)[XSP],
                                        int tid) {
  for (int i = tid; i < 64 * XN; i += TPB) {
    int p = i / XN;
    int s = i - p * XN;
    int c = p >> 3, l = p & 7;
    float d = ocN[c][s] * nnN[l][s];
    if (SUB) d -= ocO[c][s] * nnO[l][s];
    sVJS[p * XSTR + s] += d;
  }
  for (int i = tid; i < CC * XN; i += TPB) {
    int c = i / XN;
    int s = i - c * XN;
    float d = ocN[c][s];
    if (SUB) d -= ocO[c][s];
    sVJS[(64 + c) * XSTR + s] += d;
  }
}

__global__ __launch_bounds__(TPB)
void clusternet_vote_kernel(const float* __restrict__ ca,
                            const float* __restrict__ nnp,
                            float* __restrict__ out) {
  __shared__ float sVJS[NPL * XSTR];                 // 42.2 KB vertical sums (f32 exact)
  __shared__ float sOcN[CC][XSP], sNnN[LL][XSP];     // incoming row staging
  __shared__ float sOcO[CC][XSP], sNnO[LL][XSP];     // outgoing row staging
  __shared__ float sJ[NPL][SW];                      // horizontal box sums (WMMA D)
  __shared__ float sW[CC][SW];                       // center oc row, then w_c = oc_c/S_c

  const int tid  = threadIdx.x;
  const int x0   = blockIdx.x * SW;
  const int y0   = blockIdx.y * YB;
  const int lane = tid & 31;
  const int wave = tid >> 5;
  const int pt   = wave >> 1;           // plane tile 0..4 (16 planes each)
  const int xb   = (wave & 1) * 16;     // x tile base within strip
  const int Mrow = pt * 16 + (lane & 15);

  // Zero VJS so never-touched columns/planes feed exact 0 (not NaN) into WMMA.
  for (int i = tid; i < NPL * XSTR; i += TPB) sVJS[i] = 0.0f;

  // Constant banded B fragments: B_m[q=32m+k][j] = 1 iff j <= q <= j+50.
  // 16-bit B (32x16): lanes 0-15 hold K=0..7,16..23, lanes 16-31 hold K=8..15,24..31,
  // column N = lane%16, element e -> K = kbase + (e<8 ? e : e+8).
  v16h Bfrag[3];
  {
    const int j = lane & 15;
    const int kbase = (lane < 16) ? 0 : 8;
#pragma unroll
    for (int m = 0; m < 3; ++m) {
#pragma unroll
      for (int e = 0; e < 16; ++e) {
        int q = 32 * m + kbase + ((e < 8) ? e : (e + 8));
        Bfrag[m][e] = (_Float16)(((q >= j) && (q <= j + 50)) ? 1.0f : 0.0f);
      }
    }
  }
  __syncthreads();

  // Warm-up: accumulate rows [y0-RAD-1, y0+RAD-1] so the first main step
  // (add y0+RAD, sub y0-RAD-1) yields the correct 51-row window.
  for (int yy = y0 - RAD - 1; yy <= y0 + RAD - 1; ++yy) {
    load_rows(ca, nnp, yy, x0, sOcN, sNnN, tid);
    __syncthreads();
    vupdate<false>(sVJS, sOcN, sNnN, sOcO, sNnO, tid);
    __syncthreads();
  }

  for (int y = y0; y < y0 + YB; ++y) {
    // Stage incoming / outgoing rows + center oc row; prefetch next row.
    load_rows(ca, nnp, y + RAD, x0, sOcN, sNnN, tid);
    load_rows(ca, nnp, y - RAD - 1, x0, sOcO, sNnO, tid);
    for (int i = tid; i < CC * SW; i += TPB) {
      int c = i >> 5, x = i & 31;
      sW[c][x] = ca[(size_t)c * HH * WW + (size_t)y * WW + (size_t)(x0 + x)] + 1e-6f;
    }
    if (tid < 2 * CC) {
      int yy = y + RAD + 1;
      if (yy < HH) {
        size_t base = (size_t)(tid & 7) * HH * WW + (size_t)yy * WW +
                      (size_t)(x0 > RAD ? x0 - RAD : 0);
        const float* p = (tid < CC) ? (ca + base) : (nnp + base);
        __builtin_prefetch(p, 0, 1);   // -> global_prefetch_b8
      }
    }
    __syncthreads();

    // Slide the 72 vertical windows down one row (exact f32 add/sub).
    vupdate<true>(sVJS, sOcN, sNnN, sOcO, sNnO, tid);
    __syncthreads();

    // Horizontal 51-tap box sum as banded matmul: D(16 planes x 16 cols) =
    // A(planes x 96 window) * B(96 x 16), K-chunked into 3 WMMAs.
    {
      v8f acc = {};
      const float* rowp = &sVJS[Mrow * XSTR];
      const int koff = (lane < 16) ? 0 : 8;   // 16-bit A layout half-split
#pragma unroll
      for (int m = 0; m < 3; ++m) {
        v16h a;
        const int base = xb + 32 * m + koff;
#pragma unroll
        for (int e = 0; e < 16; ++e) {
          int k = (e < 8) ? e : (e + 8);      // e<8 -> K 0..7 ; e>=8 -> K 16..23
          a[e] = (_Float16)rowp[base + k];
        }
        acc = __builtin_amdgcn_wmma_f32_16x16x32_f16(
            /*neg_a=*/false, a, /*neg_b=*/false, Bfrag[m],
            /*c_mod=*/(short)0, acc, /*reuse_a=*/false, /*reuse_b=*/false);
      }
      // D layout: lane%16 = column; VGPR r -> plane row r (lanes 0-15) / r+8.
      const int x    = xb + (lane & 15);
      const int mofs = (lane < 16) ? 0 : 8;
#pragma unroll
      for (int r = 0; r < 8; ++r) sJ[pt * 16 + mofs + r][x] = acc[r];
    }
    __syncthreads();

    // Factor the weight once: w_c[x] = oc_c[x] / S_c[x].  One IEEE divide per
    // thread (256 total) instead of 8 per output pixel (2048 total).
    if (tid < CC * SW) {
      int c = tid >> 5, x = tid & 31;
      sW[c][x] = sW[c][x] / sJ[64 + c][x];
    }
    __syncthreads();

    // Per-pixel combine: out[l,x] = sum_c w_c[x] * J_{c,l}[x]  (pure FMA).
    if (tid < LL * SW) {
      int l = tid >> 5, x = tid & 31;
      float r = 0.0f;
#pragma unroll
      for (int c = 0; c < CC; ++c)
        r += sW[c][x] * sJ[c * 8 + l][x];
      out[(size_t)l * HH * WW + (size_t)y * WW + (size_t)(x0 + x)] = r;
    }
    __syncthreads();
  }
}

extern "C" void kernel_launch(void* const* d_in, const int* in_sizes, int n_in,
                              void* d_out, int out_size, void* d_ws, size_t ws_size,
                              hipStream_t stream) {
  (void)in_sizes; (void)n_in; (void)d_ws; (void)ws_size; (void)out_size;
  const float* ca  = (const float*)d_in[0];   // (8,1024,1024) cluster_assignments
  const float* nnp = (const float*)d_in[1];   // (1,8,1024,1024) nn_probs
  float* out = (float*)d_out;                 // (8,1024,1024)
  dim3 grid(WW / SW, HH / YB);                // 32 x 8 = 256 workgroups
  clusternet_vote_kernel<<<grid, dim3(TPB), 0, stream>>>(ca, nnp, out);
}
